// PointerThinkSort_33681133535785
// MI455X (gfx1250) — compile-verified, matrix operand order checked by reading, not run on hardware
//
#include <hip/hip_runtime.h>
#include <math.h>

// ---------------------------------------------------------------------------
// PointerThinkSort forward for MI455X (gfx1250, wave32, WMMA + TDM).
// bf16 operand pipeline: weights converted once, activation producers emit
// bf16 mirrors; GEMM tiles are DMA'd into LDS by the Tensor Data Mover
// (tensor_load_to_lds + s_wait_tensorcnt), double-buffered against WMMA.
// ---------------------------------------------------------------------------

#define kB   32
#define kS   512
#define kD   512
#define kH   8
#define kHD  64
#define kDFF 2048
#define kNL  4

typedef __attribute__((ext_vector_type(16))) __bf16 v16bf;
typedef __attribute__((ext_vector_type(8)))  float  v8f;
typedef __attribute__((ext_vector_type(4)))  unsigned int u32x4;
typedef __attribute__((ext_vector_type(8)))  int i32x8;
typedef __attribute__((ext_vector_type(4)))  int i32x4;

#if __has_builtin(__builtin_amdgcn_tensor_load_to_lds) && __has_builtin(__builtin_amdgcn_s_wait_tensorcnt)
#define USE_TDM 1
#else
#define USE_TDM 0
#endif

// ----------------------------- GEMM ----------------------------------------
// C/Cb[z][m][n] = act( scale * sum_k A[z][m][k] * B[z][n][k] + bias[n] )
// A: M x K rows (K contiguous, stride lda), B: N x K rows (K contiguous).
// Requires K % 32 == 0 (uses: 64, 512, 2048).
#define TM 128
#define TN 128
#define TK 32
#define LSTR 40   // LDS row stride in bf16 elems (80B: 64B data + 16B pad)

#if USE_TDM
// One TDM descriptor moves a whole (rows x 32) bf16 tile into LDS, padding
// each 64B row with 16B to produce the LSTR=40 layout. tensor_dim0/1 are the
// remaining K/rows from the tile origin so out-of-range reads return zero.
__device__ __forceinline__ void tdm_load_tile(const __bf16* gsrc, unsigned lds_byte_addr,
                                              int lda_elems, int rows_avail, int cols_avail)
{
    unsigned long long ga = (unsigned long long)(uintptr_t)gsrc;
    u32x4 g0;
    g0[0] = 1u;                                    // count=1, user descriptor
    g0[1] = lds_byte_addr;                         // lds_addr[63:32]
    g0[2] = (unsigned)(ga & 0xffffffffu);          // global_addr lo
    g0[3] = (unsigned)((ga >> 32) & 0x01ffffffu)   // global_addr hi (57-bit)
          | 0x80000000u;                           // type=2 ("image")
    i32x8 g1;
    g1[0] = (1 << 16)      // data_size = 2 bytes
          | (1 << 20)      // pad_enable
          | (3 << 22)      // pad_interval: 16 DWORDs (=64B, one tile row)
          | (3 << 25);     // pad_amount:   4 DWORDs (=16B)
    g1[1] = (int)(((unsigned)cols_avail & 0xffffu) << 16);               // tensor_dim0 lo16
    g1[2] = (int)((((unsigned)cols_avail >> 16) & 0xffffu)
          | (((unsigned)rows_avail & 0xffffu) << 16));                   // dim0 hi16 | dim1 lo16
    g1[3] = (int)((((unsigned)rows_avail >> 16) & 0xffffu)
          | ((unsigned)TK << 16));                                       // dim1 hi16 | tile_dim0=32
    g1[4] = TM;                                                          // tile_dim1=128, tile_dim2=0
    g1[5] = lda_elems;                                                   // tensor_dim0_stride lo32
    g1[6] = 0;                                                           // stride hi / dim1_stride lo
    g1[7] = 0;
    i32x4 z4 = {0, 0, 0, 0};
#if __clang_major__ >= 23
    i32x8 z8 = {0, 0, 0, 0, 0, 0, 0, 0};
    __builtin_amdgcn_tensor_load_to_lds(g0, g1, z4, z4, z8, 0);
#else
    __builtin_amdgcn_tensor_load_to_lds(g0, g1, z4, z4, 0);
#endif
}
#endif  // USE_TDM

__global__ __launch_bounds__(256)
void gemm_bf16wmma_kernel(const __bf16* __restrict__ A, const __bf16* __restrict__ Bm,
                          const float* __restrict__ bias,
                          float* __restrict__ C, __bf16* __restrict__ Cb,
                          int M, int N, int K, int lda, int ldb, int ldc,
                          long long sAz, long long sBz, long long sCz,
                          float scale, int act)
{
    __shared__ __align__(16) __bf16 As[2][TM * LSTR];
    __shared__ __align__(16) __bf16 Bs[2][TN * LSTR];

    const int z = blockIdx.z;
    A  += (long long)z * sAz;
    Bm += (long long)z * sBz;

    const int m0 = blockIdx.y * TM;
    const int n0 = blockIdx.x * TN;

    const int tid  = threadIdx.x;
    const int lane = tid & 31;
    const int wv   = tid >> 5;      // 0..7
    const int wm   = wv & 3;        // 0..3 -> 32-row strip
    const int wn   = wv >> 2;       // 0..1 -> 64-col strip
    const int hlf  = lane >> 4;     // K-half selector (this code's contiguous k = 16*hlf + s)
    const int lr   = lane & 15;

    v8f acc[2][4];
#pragma unroll
    for (int mi = 0; mi < 2; ++mi)
#pragma unroll
        for (int ni = 0; ni < 4; ++ni)
            acc[mi][ni] = (v8f){0.f,0.f,0.f,0.f,0.f,0.f,0.f,0.f};

#if USE_TDM
    // ---- TDM double-buffered pipeline ----
    if (wv == 0)
        tdm_load_tile(A + (long long)m0 * lda, (unsigned)(uintptr_t)&As[0][0],
                      lda, M - m0, K);
    if (wv == 1)
        tdm_load_tile(Bm + (long long)n0 * ldb, (unsigned)(uintptr_t)&Bs[0][0],
                      ldb, N - n0, K);

    int buf = 0;
    for (int k0 = 0; k0 < K; k0 += TK) {
        __builtin_amdgcn_s_wait_tensorcnt(0);   // all waves: safe when cnt==0
        __syncthreads();                        // tile `buf` visible to all

        const int nk = k0 + TK;
        if (nk < K) {                           // overlap next DMA with WMMA
            if (wv == 0)
                tdm_load_tile(A + (long long)m0 * lda + nk,
                              (unsigned)(uintptr_t)&As[buf ^ 1][0],
                              lda, M - m0, K - nk);
            if (wv == 1)
                tdm_load_tile(Bm + (long long)n0 * ldb + nk,
                              (unsigned)(uintptr_t)&Bs[buf ^ 1][0],
                              ldb, N - n0, K - nk);
        }

        v16bf afr[2], bfr[4];
#pragma unroll
        for (int mi = 0; mi < 2; ++mi)
            afr[mi] = *(const v16bf*)&As[buf][(wm * 32 + mi * 16 + lr) * LSTR + hlf * 16];
#pragma unroll
        for (int ni = 0; ni < 4; ++ni)
            bfr[ni] = *(const v16bf*)&Bs[buf][(wn * 64 + ni * 16 + lr) * LSTR + hlf * 16];

#pragma unroll
        for (int mi = 0; mi < 2; ++mi)
#pragma unroll
            for (int ni = 0; ni < 4; ++ni)
                acc[mi][ni] = __builtin_amdgcn_wmma_f32_16x16x32_bf16(
                    false, afr[mi], false, bfr[ni], (short)0, acc[mi][ni], false, false);

        __syncthreads();                        // reads of `buf` done before reuse
        buf ^= 1;
    }
#else
    // ---- fallback: manual global->LDS staging (bf16 copy) ----
    const int srow = tid >> 1;          // 0..127
    const int scol = (tid & 1) * 16;    // 0 or 16 (bf16 elems)
    for (int k0 = 0; k0 < K; k0 += TK) {
        {
            const int gm = m0 + srow;
            uint4 a0 = {0,0,0,0}, a1 = {0,0,0,0};
            if (gm < M) {
                const __bf16* src = A + (long long)gm * lda + k0 + scol;
                a0 = *(const uint4*)src;
                a1 = *(const uint4*)(src + 8);
            }
            *(uint4*)&As[0][srow * LSTR + scol]     = a0;
            *(uint4*)&As[0][srow * LSTR + scol + 8] = a1;
        }
        {
            const int gn = n0 + srow;
            uint4 b0 = {0,0,0,0}, b1 = {0,0,0,0};
            if (gn < N) {
                const __bf16* src = Bm + (long long)gn * ldb + k0 + scol;
                b0 = *(const uint4*)src;
                b1 = *(const uint4*)(src + 8);
            }
            *(uint4*)&Bs[0][srow * LSTR + scol]     = b0;
            *(uint4*)&Bs[0][srow * LSTR + scol + 8] = b1;
        }
        __syncthreads();
        if (k0 + TK < K && (m0 + srow) < M)
            __builtin_prefetch(A + (long long)(m0 + srow) * lda + k0 + TK + scol, 0, 1);

        v16bf afr[2], bfr[4];
#pragma unroll
        for (int mi = 0; mi < 2; ++mi)
            afr[mi] = *(const v16bf*)&As[0][(wm * 32 + mi * 16 + lr) * LSTR + hlf * 16];
#pragma unroll
        for (int ni = 0; ni < 4; ++ni)
            bfr[ni] = *(const v16bf*)&Bs[0][(wn * 64 + ni * 16 + lr) * LSTR + hlf * 16];
#pragma unroll
        for (int mi = 0; mi < 2; ++mi)
#pragma unroll
            for (int ni = 0; ni < 4; ++ni)
                acc[mi][ni] = __builtin_amdgcn_wmma_f32_16x16x32_bf16(
                    false, afr[mi], false, bfr[ni], (short)0, acc[mi][ni], false, false);
        __syncthreads();
    }
#endif

    // ---- epilogue: scale, bias, activation, fp32 and/or bf16 stores ----
    float*  Co = C  ? (C  + (long long)z * sCz) : (float*)nullptr;
    __bf16* Cbo = Cb ? (Cb + (long long)z * sCz) : (__bf16*)nullptr;
#pragma unroll
    for (int mi = 0; mi < 2; ++mi) {
#pragma unroll
        for (int ni = 0; ni < 4; ++ni) {
            const int col = n0 + wn * 64 + ni * 16 + lr;
            float bv = 0.f;
            if (bias != nullptr && col < N) bv = bias[col];
#pragma unroll
            for (int r = 0; r < 8; ++r) {
                const int row = m0 + wm * 32 + mi * 16 + r + 8 * hlf;
                if (row < M && col < N) {
                    float v = acc[mi][ni][r] * scale + bv;
                    if (act == 1) v = 0.5f * v * (1.0f + erff(v * 0.70710678118654752f));
                    const long long idx = (long long)row * ldc + col;
                    if (Co)  Co[idx]  = v;
                    if (Cbo) Cbo[idx] = (__bf16)v;
                }
            }
        }
    }
}

// ----------------------- elementwise kernels --------------------------------

__device__ __forceinline__ float wave_sum(float v) {
#pragma unroll
    for (int off = 16; off > 0; off >>= 1) v += __shfl_down(v, off, 32);
    return v;
}
__device__ __forceinline__ float wave_max(float v) {
#pragma unroll
    for (int off = 16; off > 0; off >>= 1) v = fmaxf(v, __shfl_down(v, off, 32));
    return v;
}

// f32 -> bf16 bulk convert (weights, step_embed)
__global__ void cvt_bf16_kernel(const float* __restrict__ src, __bf16* __restrict__ dst,
                                long long n)
{
    const long long i = (long long)blockIdx.x * 256 + threadIdx.x;
    if (i < n) dst[i] = (__bf16)src[i];
}

// h[t] = token_embed[x[t]] + pos_embed[t % S]  (fp32 + bf16 mirror)
__global__ __launch_bounds__(128)
void embed_kernel(const int* __restrict__ x, const float* __restrict__ tok,
                  const float* __restrict__ pos, float* __restrict__ h,
                  __bf16* __restrict__ hb)
{
    const int t = blockIdx.x;           // 0..B*S-1
    const int s = t & (kS - 1);
    const int id = x[t];
    const int c = threadIdx.x;          // 0..127: one float4 of D=512
    const float4 a = ((const float4*)(tok + (long long)id * kD))[c];
    const float4 p = ((const float4*)(pos + (long long)s  * kD))[c];
    float4 o; o.x = a.x + p.x; o.y = a.y + p.y; o.z = a.z + p.z; o.w = a.w + p.w;
    ((float4*)(h + (long long)t * kD))[c] = o;
    __bf16* hbr = hb + (long long)t * kD + c * 4;
    hbr[0] = (__bf16)o.x; hbr[1] = (__bf16)o.y; hbr[2] = (__bf16)o.z; hbr[3] = (__bf16)o.w;
}

// Y/Yb[row] = LN(X[row] + R[row]) * g + b   (R, Y, Yb each optional)
__global__ __launch_bounds__(128)
void ln_kernel(const float* __restrict__ X, const float* __restrict__ R,
               const float* __restrict__ g, const float* __restrict__ b,
               float* __restrict__ Y, __bf16* __restrict__ Yb)
{
    __shared__ float sb[4];
    const long long row = blockIdx.x;
    const float* xr = X + row * kD;
    const float* rr = (R != nullptr) ? (R + row * kD) : nullptr;
    const int t = threadIdx.x;

    float v[4]; float s = 0.f;
#pragma unroll
    for (int i = 0; i < 4; ++i) {
        const int c = t + 128 * i;
        float u = xr[c] + (rr ? rr[c] : 0.f);
        v[i] = u; s += u;
    }
    s = wave_sum(s);
    if ((t & 31) == 0) sb[t >> 5] = s;
    __syncthreads();
    const float mean = (sb[0] + sb[1] + sb[2] + sb[3]) * (1.0f / kD);
    __syncthreads();

    float s2 = 0.f;
#pragma unroll
    for (int i = 0; i < 4; ++i) { const float d = v[i] - mean; s2 += d * d; }
    s2 = wave_sum(s2);
    if ((t & 31) == 0) sb[t >> 5] = s2;
    __syncthreads();
    const float var = (sb[0] + sb[1] + sb[2] + sb[3]) * (1.0f / kD);
    const float rstd = rsqrtf(var + 1e-5f);
#pragma unroll
    for (int i = 0; i < 4; ++i) {
        const int c = t + 128 * i;
        const float y = (v[i] - mean) * rstd * g[c] + b[c];
        if (Y)  Y [row * kD + c] = y;
        if (Yb) Yb[row * kD + c] = (__bf16)y;
    }
}

// softmax over rows of 512: fp32 in -> bf16 probs out
__global__ __launch_bounds__(128)
void softmax_kernel(const float* __restrict__ P, __bf16* __restrict__ O)
{
    __shared__ float sb[4];
    const float* p = P + (long long)blockIdx.x * kS;
    __bf16* o = O + (long long)blockIdx.x * kS;
    const int t = threadIdx.x;

    float v[4]; float mx = -3.0e38f;
#pragma unroll
    for (int i = 0; i < 4; ++i) { v[i] = p[t + 128 * i]; mx = fmaxf(mx, v[i]); }
    mx = wave_max(mx);
    if ((t & 31) == 0) sb[t >> 5] = mx;
    __syncthreads();
    mx = fmaxf(fmaxf(sb[0], sb[1]), fmaxf(sb[2], sb[3]));
    __syncthreads();

    float s = 0.f;
#pragma unroll
    for (int i = 0; i < 4; ++i) { v[i] = __expf(v[i] - mx); s += v[i]; }
    s = wave_sum(s);
    if ((t & 31) == 0) sb[t >> 5] = s;
    __syncthreads();
    const float inv = 1.0f / (sb[0] + sb[1] + sb[2] + sb[3]);
#pragma unroll
    for (int i = 0; i < 4; ++i) o[t + 128 * i] = (__bf16)(v[i] * inv);
}

// vt[b][h][d][l] = qkv_bf16[b*S+l][2*D + h*64 + d]
__global__ void vtrans_kernel(const __bf16* __restrict__ qkv, __bf16* __restrict__ vt)
{
    long long i = (long long)blockIdx.x * 256 + threadIdx.x;
    const long long total = (long long)kB * kH * kHD * kS;
    if (i >= total) return;
    const int l = (int)(i & (kS - 1)); long long j = i >> 9;
    const int d = (int)(j & (kHD - 1)); j >>= 6;
    const int h = (int)(j & (kH - 1));
    const int b = (int)(j >> 3);
    vt[i] = qkv[((long long)(b * kS + l)) * (3 * kD) + 2 * kD + h * kHD + d];
}

// out_bf16[i] = enc_f32[i] + sel[i % 512]
__global__ void addsel_kernel(const float* __restrict__ enc, const float* __restrict__ sel,
                              __bf16* __restrict__ out, long long n)
{
    const long long i = (long long)blockIdx.x * 256 + threadIdx.x;
    if (i < n) out[i] = (__bf16)(enc[i] + sel[i & (kD - 1)]);
}

// ----------------------------- host side ------------------------------------

extern "C" void kernel_launch(void* const* d_in, const int* in_sizes, int n_in,
                              void* d_out, int out_size, void* d_ws, size_t ws_size,
                              hipStream_t stream)
{
    (void)in_sizes; (void)n_in; (void)out_size; (void)ws_size;

    const int*   x      = (const int*)  d_in[0];
    const float* tok    = (const float*)d_in[1];
    const float* pos    = (const float*)d_in[2];
    const float* stepe  = (const float*)d_in[3];
    const float* sel    = (const float*)d_in[4];
    const float* qkv_w  = (const float*)d_in[5];
    const float* qkv_b  = (const float*)d_in[6];
    const float* out_w  = (const float*)d_in[7];
    const float* out_b  = (const float*)d_in[8];
    const float* w1     = (const float*)d_in[9];
    const float* b1     = (const float*)d_in[10];
    const float* w2     = (const float*)d_in[11];
    const float* b2     = (const float*)d_in[12];
    const float* ln1g   = (const float*)d_in[13];
    const float* ln1b   = (const float*)d_in[14];
    const float* ln2g   = (const float*)d_in[15];
    const float* ln2b   = (const float*)d_in[16];
    const float* encg   = (const float*)d_in[17];
    const float* encb   = (const float*)d_in[18];
    const float* pq_w   = (const float*)d_in[19];
    const float* pq_b   = (const float*)d_in[20];
    const float* pk_w   = (const float*)d_in[21];
    const float* pk_b   = (const float*)d_in[22];

    // workspace layout (assumes ws_size >= ~247 MB)
    char* ws = (char*)d_ws;
    float*  Wh   = (float*) (ws);                          // h fp32        32 MB
    __bf16* Whb  = (__bf16*)(ws + (size_t)33554432);       // h bf16        16 MB
    __bf16* Wbig = (__bf16*)(ws + (size_t)50331648);       // qkv/ff1 bf16  64 MB
    float*  Wt   = (float*) (ws + (size_t)117440512);      // o_proj/ff2    32 MB
    __bf16* Wob  = (__bf16*)(ws + (size_t)150994944);      // attn o bf16   16 MB
    float*  Wsc  = (float*) (ws + (size_t)167772160);      // scores fp32    8 MB
    __bf16* Wscb = (__bf16*)(ws + (size_t)176160768);      // probs bf16     4 MB
    __bf16* Wvtb = (__bf16*)(ws + (size_t)180355072);      // Vt bf16       16 MB
    __bf16* Wqb  = (__bf16*)(ws + (size_t)197132288);      // q bf16       0.5 MB
    __bf16* Wkb  = (__bf16*)(ws + (size_t)197656576);      // k bf16        16 MB
    __bf16* Wksb = (__bf16*)(ws + (size_t)214433792);      // enc+sel bf16  16 MB
    __bf16* qkv_wb = (__bf16*)(ws + (size_t)231211008);    // 6 MB
    __bf16* out_wb = (__bf16*)(ws + (size_t)237502464);    // 2 MB
    __bf16* w1b    = (__bf16*)(ws + (size_t)239599616);    // 8 MB
    __bf16* w2b    = (__bf16*)(ws + (size_t)247988224);    // 8 MB
    __bf16* pq_wb  = (__bf16*)(ws + (size_t)256376832);    // 0.5 MB
    __bf16* pk_wb  = (__bf16*)(ws + (size_t)256901120);    // 0.5 MB
    __bf16* stepb  = (__bf16*)(ws + (size_t)257425408);    // 0.5 MB

    float* logits = (float*)d_out;                          // B*S*S
    float* enc    = logits + (size_t)kB * kS * kS;          // B*S*D

    auto cvt = [&](const float* s, __bf16* d, long long n) {
        cvt_bf16_kernel<<<(unsigned)((n + 255) / 256), 256, 0, stream>>>(s, d, n);
    };
    auto gemm = [&](const __bf16* A, const __bf16* Bm, const float* bias,
                    float* C, __bf16* Cb,
                    int M, int N, int K, int lda, int ldb, int ldc,
                    long long sAz, long long sBz, long long sCz, int Z,
                    float scale, int act) {
        dim3 grid((N + TN - 1) / TN, (M + TM - 1) / TM, Z);
        gemm_bf16wmma_kernel<<<grid, 256, 0, stream>>>(
            A, Bm, bias, C, Cb, M, N, K, lda, ldb, ldc, sAz, sBz, sCz, scale, act);
    };

    const int NT = kB * kS;  // 16384 tokens

    // one-time weight conversions
    cvt(qkv_w, qkv_wb, (long long)kNL * 3 * kD * kD);
    cvt(out_w, out_wb, (long long)kNL * kD * kD);
    cvt(w1,    w1b,    (long long)kNL * kDFF * kD);
    cvt(w2,    w2b,    (long long)kNL * kD * kDFF);
    cvt(pq_w,  pq_wb,  (long long)kD * kD);
    cvt(pk_w,  pk_wb,  (long long)kD * kD);
    cvt(stepe, stepb,  (long long)kS * kD);

    embed_kernel<<<NT, 128, 0, stream>>>(x, tok, pos, Wh, Whb);

    for (int li = 0; li < kNL; ++li) {
        // QKV projection (bf16 out only): (16384 x 1536)
        gemm(Whb, qkv_wb + (size_t)li * 3 * kD * kD, qkv_b + (size_t)li * 3 * kD,
             nullptr, Wbig, NT, 3 * kD, kD, kD, kD, 3 * kD, 0, 0, 0, 1, 1.0f, 0);

        // Transposed V (bf16) for attn@V as A·B^T
        vtrans_kernel<<<((kB * kH * kHD * kS) + 255) / 256, 256, 0, stream>>>(Wbig, Wvtb);

        for (int b = 0; b < kB; ++b) {
            const __bf16* qb = Wbig + (size_t)b * kS * 3 * kD;
            // scores[h][s][l] = q·k^T / 8 per head (grid.z = H), fp32 out
            gemm(qb, qb + kD, nullptr, Wsc, nullptr,
                 kS, kS, kHD, 3 * kD, 3 * kD, kS,
                 kHD, kHD, (long long)kS * kS, kH, 0.125f, 0);
            softmax_kernel<<<kH * kS, 128, 0, stream>>>(Wsc, Wscb);
            // o[s][h*64+d] = probs @ Vt^T per head, bf16 out
            gemm(Wscb, Wvtb + (size_t)b * kH * kHD * kS, nullptr,
                 nullptr, Wob + (size_t)b * kS * kD,
                 kS, kHD, kS, kS, kS, kD,
                 (long long)kS * kS, (long long)kHD * kS, kHD, kH, 1.0f, 0);
        }

        // out projection (fp32 out, residual input for LN)
        gemm(Wob, out_wb + (size_t)li * kD * kD, out_b + (size_t)li * kD,
             Wt, nullptr, NT, kD, kD, kD, kD, kD, 0, 0, 0, 1, 1.0f, 0);
        // h = LN(h + o_proj): fp32 + bf16 mirror
        ln_kernel<<<NT, 128, 0, stream>>>(Wh, Wt, ln1g + (size_t)li * kD,
                                          ln1b + (size_t)li * kD, Wh, Whb);
        // ff1 = gelu(h @ w1^T + b1): bf16 out
        gemm(Whb, w1b + (size_t)li * kDFF * kD, b1 + (size_t)li * kDFF,
             nullptr, Wbig, NT, kDFF, kD, kD, kD, kDFF, 0, 0, 0, 1, 1.0f, 1);
        // ff2 = ff1 @ w2^T + b2: fp32 out
        gemm(Wbig, w2b + (size_t)li * kD * kDFF, b2 + (size_t)li * kD,
             Wt, nullptr, NT, kD, kDFF, kDFF, kDFF, kD, 0, 0, 0, 1, 1.0f, 0);
        // h = LN(h + ff2)
        ln_kernel<<<NT, 128, 0, stream>>>(Wh, Wt, ln2g + (size_t)li * kD,
                                          ln2b + (size_t)li * kD, Wh, Whb);
    }

    // enc = LN(h) -> second output (fp32 only)
    ln_kernel<<<NT, 128, 0, stream>>>(Wh, nullptr, encg, encb, enc, nullptr);

    // k-source = enc + sel_embed[0] (bf16)
    addsel_kernel<<<(int)(((long long)NT * kD + 255) / 256), 256, 0, stream>>>(
        enc, sel, Wksb, (long long)NT * kD);

    // q = step_embed @ pq_w^T + pq_b (bf16 out)
    gemm(stepb, pq_wb, pq_b, nullptr, Wqb, kS, kD, kD, kD, kD, kD, 0, 0, 0, 1, 1.0f, 0);
    // k = (enc + sel0) @ pk_w^T + pk_b (bf16 out)
    gemm(Wksb, pk_wb, pk_b, nullptr, Wkb, NT, kD, kD, kD, kD, kD, 0, 0, 0, 1, 1.0f, 0);

    // logits[b][s][l] = q[s] · k[b][l] / sqrt(D)  (fp32 -> d_out)
    const float lsc = 0.04419417382415922f;  // 1/sqrt(512)
    for (int b = 0; b < kB; ++b) {
        gemm(Wqb, Wkb + (size_t)b * kS * kD, nullptr,
             logits + (size_t)b * kS * kS, nullptr,
             kS, kS, kD, kD, kD, kS, 0, 0, 0, 1, lsc, 0);
    }
}
